// TransformerLayer_9397388444090
// MI455X (gfx1250) — matrix operand order repaired and checked by reading, not compiled, not run on hardware
//
#include <hip/hip_runtime.h>
#include <math.h>

typedef _Float16 f16;
typedef __attribute__((ext_vector_type(16))) _Float16 v16h;
typedef __attribute__((ext_vector_type(8)))  float    v8f;

// ---------------------------------------------------------------------------
// WMMA helpers (CDNA5 gfx1250, wave32).
// A layout (16x32 f16, doc 7.12.2): lane<16 holds row M=lane, halves = K
// {0..7,16..23}; lane>=16 holds M=lane-16, halves = K {8..15,24..31}.
// B layout (32x16 f16): half hh holds K=(hh<8 ? hh : hh+8) (lanes 0-15, N=lane) and
// K=8+(hh<8 ? hh : hh+8) (lanes 16-31, N=lane-16)  => half hh maps to K=koff+(hh<8 ? hh : hh+8), as A.
// C/D layout (16x16 f32): VGPR r -> M=r+8*(lane>=16), N=lane&15.
// ---------------------------------------------------------------------------
__device__ __forceinline__ v16h load_a(const f16* __restrict__ base, int strideH,
                                       int kc, int lane) {
  int row  = lane & 15;
  int koff = kc * 32 + ((lane & 16) ? 8 : 0);
  const f16* p = base + row * strideH + koff;
  union { v16h v; uint4 q[2]; } u;
  u.q[0] = *(const uint4*)(p);        // halves: K = koff .. koff+7
  u.q[1] = *(const uint4*)(p + 16);   // halves: K = koff+16 .. koff+23
  return u.v;
}

// Weights are pre-swizzled: tile = nc*numKc + kc, 512 halves per tile,
// lane-major (16 contiguous halves per lane) -> single 32B load.
__device__ __forceinline__ v16h load_b(const f16* __restrict__ pk, int tile, int lane) {
  return *(const v16h*)(pk + (size_t)tile * 512 + (size_t)lane * 16);
}

__device__ __forceinline__ v8f wmma16(v16h a, v16h b, v8f c) {
  return __builtin_amdgcn_wmma_f32_16x16x32_f16(false, a, false, b, (short)0, c,
                                                false, false);
}

// Async global->LDS copy of 16 bytes per lane (CDNA5 VGLOBAL opcode 98,
// tracked with ASYNCcnt). ldsoff = LDS byte offset (low 32 bits of generic
// shared pointer, per LDS aperture truncation rule), gaddr = 64-bit global.
__device__ __forceinline__ void async_copy_b128(unsigned ldsoff,
                                                unsigned long long gaddr) {
  asm volatile("global_load_async_to_lds_b128 %0, %1, off"
               :: "v"(ldsoff), "v"(gaddr) : "memory");
}
__device__ __forceinline__ void wait_asynccnt0() {
  asm volatile("s_wait_asynccnt 0x0" ::: "memory");
}

// ---------------------------------------------------------------------------
// Pack a logical W'[K][N] f32 weight into swizzled f16 B-operand tiles.
// mode 0: src row-major [K][Ncols].  mode 1: per-head src[(h*K + k)*64 + d],
// with n = h*64+d (stacked-head column).
// ---------------------------------------------------------------------------
__global__ void pack_weight_kernel(const float* __restrict__ src, f16* __restrict__ dst,
                                   int K, int Ncols, int numKc, int mode, int total) {
  int idx = blockIdx.x * blockDim.x + threadIdx.x;
  if (idx >= total) return;
  int hh   = idx & 15;
  int lane = (idx >> 4) & 31;
  int t    = idx >> 9;
  int kc   = t % numKc;
  int nc   = t / numKc;
  int k = kc * 32 + ((lane & 16) ? 8 : 0) + (hh < 8 ? hh : hh + 8);
  int n = nc * 16 + (lane & 15);
  float v;
  if (mode == 0) {
    v = src[k * Ncols + n];
  } else {
    int h = n >> 6, d = n & 63;
    v = src[(h * K + k) * 64 + d];
  }
  dst[idx] = (f16)v;
}

__global__ void f32_to_f16_kernel(const float* __restrict__ src, f16* __restrict__ dst,
                                  int n) {
  int i = blockIdx.x * blockDim.x + threadIdx.x;
  if (i < n) dst[i] = (f16)src[i];
}

// ---------------------------------------------------------------------------
// QKV: [192,128] x [128,512] (x3 matrices). 36 blocks (3 mats x 12 m-tiles).
// ---------------------------------------------------------------------------
__global__ void qkv_kernel(const f16* __restrict__ nodes16,
                           const f16* Wqp, const f16* Wkp, const f16* Wvp,
                           const float* bq, const float* bk, const float* bv,
                           float* q, float* k, float* v) {
  __shared__ f16 Ash[16 * 128];
  int blk = blockIdx.x;
  int mat = blk / 12, mt = blk % 12;
  const f16*   W    = (mat == 0) ? Wqp : ((mat == 1) ? Wkp : Wvp);
  const float* bias = (mat == 0) ? bq  : ((mat == 1) ? bk  : bv);
  float*       dst  = (mat == 0) ? q   : ((mat == 1) ? k   : v);

  int tid = threadIdx.x;
  ((uint4*)Ash)[tid] = ((const uint4*)(nodes16 + mt * 2048))[tid];  // 16x128 f16
  __syncthreads();

  int w = tid >> 5, lane = tid & 31;
  int nlo = lane & 15, mhalf = (lane & 16) ? 8 : 0;
#pragma unroll
  for (int ct4 = 0; ct4 < 4; ct4++) {
    int nc = w * 4 + ct4;
    v8f acc = {};
#pragma unroll
    for (int kc = 0; kc < 4; kc++) {
      v16h a  = load_a(Ash, 128, kc, lane);
      v16h bb = load_b(W, nc * 4 + kc, lane);
      acc = wmma16(a, bb, acc);
    }
    int col = nc * 16 + nlo;
    float bs = bias[col];
#pragma unroll
    for (int r = 0; r < 8; r++)
      dst[(mt * 16 + mhalf + r) * 512 + col] = acc[r] + bs;
  }
}

// ---------------------------------------------------------------------------
// Fused edge kernel: one workgroup per (b,n); wave w = head h.
// ---------------------------------------------------------------------------
__global__ void edge_kernel(const f16* __restrict__ edges16,
                            const float* __restrict__ qws,
                            const float* __restrict__ kws,
                            const float* __restrict__ vws,
                            const f16* Wf1p, const f16* Wf2p,
                            const float* bf1, const float* bf2,
                            const f16* Wep,  const float* be,
                            const f16* We1p, const float* be1,
                            const f16* We2p, const float* be2,
                            float* __restrict__ wvws, float* __restrict__ outE) {
  extern __shared__ char smem[];
  float* qs  = (float*)(smem);                          //  512 f32 : 2048 B
  f16*   Esh = (f16*)(smem + 2048);                     // 96x32 f16: 6144 B
  f16*   EA  = (f16*)(smem + 2048 + 6144);              // 96x512  : 98304 B
  f16*   R1  = (f16*)(smem + 2048 + 6144 + 98304);      // 96x32   : 6144 B
  f16*   R2  = (f16*)(smem + 2048 + 6144 + 98304 + 6144);

  int blk = blockIdx.x;
  int b = blk / 96, n = blk % 96;
  int rowbase = b * 96 + n;
  int tid = threadIdx.x;

  // ---- P0: stage E tile (f16) and q row (f32) via async global->LDS ----
  {
    unsigned ldsE = (unsigned)(uintptr_t)(void*)Esh;    // 6144 B = 384 x 16B
    unsigned ldsQ = (unsigned)(uintptr_t)(void*)qs;     // 2048 B = 128 x 16B
    const f16*   gE = edges16 + (size_t)rowbase * 96 * 32;
    const float* gQ = qws + (size_t)rowbase * 512;
    for (int i = tid; i < 512; i += 256) {
      unsigned dst;
      unsigned long long src;
      if (i < 384) {
        dst = ldsE + (unsigned)i * 16u;
        src = (unsigned long long)(uintptr_t)(gE + i * 8);
      } else {
        int j = i - 384;
        dst = ldsQ + (unsigned)j * 16u;
        src = (unsigned long long)(uintptr_t)(gQ + j * 4);
      }
      async_copy_b128(dst, src);
    }
    wait_asynccnt0();
  }
  __syncthreads();

  int w = tid >> 5, lane = tid & 31;
  int nlo = lane & 15, mhalf = (lane & 16) ? 8 : 0;
  const float scale = 0.125f;  // 1/sqrt(64)

  float wvp[4][8];
#pragma unroll
  for (int c = 0; c < 4; c++)
#pragma unroll
    for (int r = 0; r < 8; r++) wvp[c][r] = 0.f;

  // ---- P1: FiLM + attn + softmax(feature dim) + wv accumulation ----
  for (int t = 0; t < 6; t++) {
    v16h a = load_a(Esh + t * 16 * 32, 32, 0, lane);
    float ea[4][8];
#pragma unroll
    for (int ct4 = 0; ct4 < 4; ct4++) {
      int nc = w * 4 + ct4;
      v16h b1 = load_b(Wf1p, nc, lane);
      v16h b2 = load_b(Wf2p, nc, lane);
      v8f z = {};
      v8f mulC = wmma16(a, b1, z);
      v8f addC = wmma16(a, b2, z);
      int col = nc * 16 + nlo;
      float qb = qs[col], b1c = bf1[col], b2c = bf2[col];
#pragma unroll
      for (int r = 0; r < 8; r++) {
        int mrow = t * 16 + mhalf + r;
        float attn = qb * kws[((size_t)(b * 96 + mrow)) * 512 + col] * scale;
        float e = (mulC[r] + b1c) * attn + (addC[r] + b2c) + attn;
        ea[ct4][r] = e;
        EA[mrow * 512 + col] = (f16)e;  // pre-softmax edge_att (for edge head-concat)
      }
    }
    // softmax over this head's 64 features per row (half-wave reductions)
#pragma unroll
    for (int r = 0; r < 8; r++) {
      float m = ea[0][r];
#pragma unroll
      for (int c = 1; c < 4; c++) m = fmaxf(m, ea[c][r]);
#pragma unroll
      for (int off = 1; off < 16; off <<= 1) m = fmaxf(m, __shfl_xor(m, off, 32));
      float ex[4], s = 0.f;
#pragma unroll
      for (int c = 0; c < 4; c++) { ex[c] = __expf(ea[c][r] - m); s += ex[c]; }
#pragma unroll
      for (int off = 1; off < 16; off <<= 1) s += __shfl_xor(s, off, 32);
      float inv = 1.0f / s;
      int mrow = t * 16 + mhalf + r;
#pragma unroll
      for (int c = 0; c < 4; c++) {
        int col = (w * 4 + c) * 16 + nlo;
        wvp[c][r] += ex[c] * inv * vws[((size_t)(b * 96 + mrow)) * 512 + col];
      }
    }
  }
  // wv reduce over rows (8 regs + partner half-wave) and store
#pragma unroll
  for (int c = 0; c < 4; c++) {
    float s = 0.f;
#pragma unroll
    for (int r = 0; r < 8; r++) s += wvp[c][r];
    s += __shfl_xor(s, 16, 32);
    if (lane < 16) wvws[(size_t)rowbase * 512 + (w * 4 + c) * 16 + nlo] = s;
  }
  __syncthreads();

  // ---- P2: edges head-concat GEMM [96,512]x[512,32] + bias + relu ----
  for (int j = w; j < 12; j += 8) {          // wave-uniform
    int t = j >> 1, ct = j & 1;
    v8f acc = {};
#pragma unroll
    for (int kc = 0; kc < 16; kc++) {
      v16h a  = load_a(EA + t * 16 * 512, 512, kc, lane);
      v16h bb = load_b(Wep, ct * 16 + kc, lane);
      acc = wmma16(a, bb, acc);
    }
    int col = ct * 16 + nlo;
    float bs = be[col];
#pragma unroll
    for (int r = 0; r < 8; r++)
      R1[(t * 16 + mhalf + r) * 32 + col] = (f16)fmaxf(acc[r] + bs, 0.f);
  }
  __syncthreads();

  // ---- P3: MLP1 [96,32]x[32,32] + relu ----
  for (int j = w; j < 12; j += 8) {
    int t = j >> 1, ct = j & 1;
    v16h a  = load_a(R1 + t * 16 * 32, 32, 0, lane);
    v16h bb = load_b(We1p, ct, lane);
    v8f z = {};
    v8f acc = wmma16(a, bb, z);
    int col = ct * 16 + nlo;
    float bs = be1[col];
#pragma unroll
    for (int r = 0; r < 8; r++)
      R2[(t * 16 + mhalf + r) * 32 + col] = (f16)fmaxf(acc[r] + bs, 0.f);
  }
  __syncthreads();

  // ---- P4: MLP2 [96,32]x[32,32] + bias, write new_edges ----
  for (int j = w; j < 12; j += 8) {
    int t = j >> 1, ct = j & 1;
    v16h a  = load_a(R2 + t * 16 * 32, 32, 0, lane);
    v16h bb = load_b(We2p, ct, lane);
    v8f z = {};
    v8f acc = wmma16(a, bb, z);
    int col = ct * 16 + nlo;
    float bs = be2[col];
#pragma unroll
    for (int r = 0; r < 8; r++) {
      int mrow = t * 16 + mhalf + r;
      outE[((size_t)rowbase * 96 + mrow) * 32 + col] = acc[r] + bs;
    }
  }
}

// ---------------------------------------------------------------------------
// Node output: [192,512]x[512,64] -> relu -> [64,64] -> relu -> [64,64]
// ---------------------------------------------------------------------------
__global__ void nodes_kernel(const float* __restrict__ wvws,
                             const f16* Wnp,  const float* bn,
                             const f16* Wn1p, const float* bn1,
                             const f16* Wn2p, const float* bn2,
                             float* __restrict__ outN) {
  __shared__ f16 A1[16 * 512];
  __shared__ f16 A2[16 * 64];
  __shared__ f16 A3[16 * 64];
  int mt = blockIdx.x;
  int tid = threadIdx.x;
  for (int i = tid; i < 16 * 512; i += 256) A1[i] = (f16)wvws[(size_t)mt * 8192 + i];
  __syncthreads();
  int w = tid >> 5, lane = tid & 31;
  int nlo = lane & 15, mhalf = (lane & 16) ? 8 : 0;

  if (w < 4) {  // wave-uniform
    int ct = w;
    v8f acc = {};
#pragma unroll
    for (int kc = 0; kc < 16; kc++) {
      v16h a  = load_a(A1, 512, kc, lane);
      v16h bb = load_b(Wnp, ct * 16 + kc, lane);
      acc = wmma16(a, bb, acc);
    }
    int col = ct * 16 + nlo;
    float bs = bn[col];
#pragma unroll
    for (int r = 0; r < 8; r++)
      A2[(mhalf + r) * 64 + col] = (f16)fmaxf(acc[r] + bs, 0.f);
  }
  __syncthreads();
  if (w < 4) {
    int ct = w;
    v8f acc = {};
#pragma unroll
    for (int kc = 0; kc < 2; kc++) {
      v16h a  = load_a(A2, 64, kc, lane);
      v16h bb = load_b(Wn1p, ct * 2 + kc, lane);
      acc = wmma16(a, bb, acc);
    }
    int col = ct * 16 + nlo;
    float bs = bn1[col];
#pragma unroll
    for (int r = 0; r < 8; r++)
      A3[(mhalf + r) * 64 + col] = (f16)fmaxf(acc[r] + bs, 0.f);
  }
  __syncthreads();
  if (w < 4) {
    int ct = w;
    v8f acc = {};
#pragma unroll
    for (int kc = 0; kc < 2; kc++) {
      v16h a  = load_a(A3, 64, kc, lane);
      v16h bb = load_b(Wn2p, ct * 2 + kc, lane);
      acc = wmma16(a, bb, acc);
    }
    int col = ct * 16 + nlo;
    float bs = bn2[col];
#pragma unroll
    for (int r = 0; r < 8; r++)
      outN[(mt * 16 + mhalf + r) * 64 + col] = acc[r] + bs;
  }
}

// ---------------------------------------------------------------------------
extern "C" void kernel_launch(void* const* d_in, const int* in_sizes, int n_in,
                              void* d_out, int out_size, void* d_ws, size_t ws_size,
                              hipStream_t stream) {
  (void)in_sizes; (void)n_in; (void)out_size; (void)ws_size;
  const float* nodes = (const float*)d_in[0];
  const float* edges = (const float*)d_in[1];
  const float* Wq  = (const float*)d_in[2];  const float* bq  = (const float*)d_in[3];
  const float* Wk  = (const float*)d_in[4];  const float* bk  = (const float*)d_in[5];
  const float* Wv  = (const float*)d_in[6];  const float* bv  = (const float*)d_in[7];
  const float* Wf1 = (const float*)d_in[8];  const float* bf1 = (const float*)d_in[9];
  const float* Wf2 = (const float*)d_in[10]; const float* bf2 = (const float*)d_in[11];
  const float* Wn  = (const float*)d_in[12]; const float* bn  = (const float*)d_in[13];
  const float* We  = (const float*)d_in[14]; const float* be  = (const float*)d_in[15];
  const float* Wn1 = (const float*)d_in[16]; const float* bn1 = (const float*)d_in[17];
  const float* Wn2 = (const float*)d_in[18]; const float* bn2 = (const float*)d_in[19];
  const float* We1 = (const float*)d_in[20]; const float* be1 = (const float*)d_in[21];
  const float* We2 = (const float*)d_in[22]; const float* be2 = (const float*)d_in[23];

  char* ws = (char*)d_ws;
  f16*   nodes16 = (f16*)(ws + 0);          // 49152 B
  f16*   Wqp  = (f16*)(ws + 49152);         // 131072
  f16*   Wkp  = (f16*)(ws + 180224);        // 131072
  f16*   Wvp  = (f16*)(ws + 311296);        // 131072
  f16*   Wf1p = (f16*)(ws + 442368);        // 32768
  f16*   Wf2p = (f16*)(ws + 475136);        // 32768
  f16*   Wep  = (f16*)(ws + 507904);        // 32768
  f16*   We1p = (f16*)(ws + 540672);        // 2048
  f16*   We2p = (f16*)(ws + 542720);        // 2048
  f16*   Wnp  = (f16*)(ws + 544768);        // 65536
  f16*   Wn1p = (f16*)(ws + 610304);        // 8192
  f16*   Wn2p = (f16*)(ws + 618496);        // 8192
  float* qws  = (float*)(ws + 626688);      // 393216
  float* kws  = (float*)(ws + 1019904);     // 393216
  float* vws  = (float*)(ws + 1413120);     // 393216
  float* wvws = (float*)(ws + 1806336);     // 393216
  f16*   edges16 = (f16*)(ws + 2199552);    // 1179648  (total ~3.4 MB)

  float* outN = (float*)d_out;
  float* outE = outN + 2 * 96 * 64;

  f32_to_f16_kernel<<<96, 256, 0, stream>>>(nodes, nodes16, 24576);
  f32_to_f16_kernel<<<2304, 256, 0, stream>>>(edges, edges16, 589824);
  pack_weight_kernel<<<256, 256, 0, stream>>>(Wq,  Wqp, 128, 512,  4, 1, 65536);
  pack_weight_kernel<<<256, 256, 0, stream>>>(Wk,  Wkp, 128, 512,  4, 1, 65536);
  pack_weight_kernel<<<256, 256, 0, stream>>>(Wv,  Wvp, 128, 512,  4, 1, 65536);
  pack_weight_kernel<<<64,  256, 0, stream>>>(Wf1, Wf1p, 32, 512,  1, 1, 16384);
  pack_weight_kernel<<<64,  256, 0, stream>>>(Wf2, Wf2p, 32, 512,  1, 1, 16384);
  pack_weight_kernel<<<64,  256, 0, stream>>>(We,  Wep, 512,  32, 16, 0, 16384);
  pack_weight_kernel<<<4,   256, 0, stream>>>(We1, We1p, 32,  32,  1, 0, 1024);
  pack_weight_kernel<<<4,   256, 0, stream>>>(We2, We2p, 32,  32,  1, 0, 1024);
  pack_weight_kernel<<<128, 256, 0, stream>>>(Wn,  Wnp, 512,  64, 16, 0, 32768);
  pack_weight_kernel<<<16,  256, 0, stream>>>(Wn1, Wn1p, 64,  64,  2, 0, 4096);
  pack_weight_kernel<<<16,  256, 0, stream>>>(Wn2, Wn2p, 64,  64,  2, 0, 4096);

  qkv_kernel<<<36, 256, 0, stream>>>(nodes16, Wqp, Wkp, Wvp, bq, bk, bv,
                                     qws, kws, vws);

  size_t smem = 2048 + 6144 + 98304 + 6144 + 6144;  // 118784 B dynamic LDS
  edge_kernel<<<192, 256, smem, stream>>>(edges16, qws, kws, vws,
                                          Wf1p, Wf2p, bf1, bf2,
                                          Wep, be, We1p, be1, We2p, be2,
                                          wvws, outE);

  nodes_kernel<<<12, 256, 0, stream>>>(wvws, Wnp, bn, Wn1p, bn1, Wn2p, bn2, outN);
}
